// BinaryXnorExceptOutliersLinear_82609400971864
// MI455X (gfx1250) — matrix operand order repaired and checked by reading, not compiled, not run on hardware
//
#include <hip/hip_runtime.h>

// ---------------------------------------------------------------------------
// BinaryXnorExceptOutliersLinear on MI455X (gfx1250)
//   y[M,N] = x[M,K] @ w_sim[N,K]^T + bias,  w_sim = mask ? w : sign(w)*scale
//   M=8192, N=11008, K=4096
//
// Phase 1: one-pass convert x -> bf16 [M][K] and w_sim -> bf16 TRANSPOSED
//          [K][N] into workspace (157 MB: fits MI455X 192 MB L2).
// Phase 2: bf16 WMMA GEMM (V_WMMA_F32_16X16X32_BF16), double-buffered LDS
//          staged via GLOBAL_LOAD_ASYNC_TO_LDS_B128 (ASYNCcnt) when available.
// Fallback: fused single kernel if ws_size is too small.
// ---------------------------------------------------------------------------

typedef __attribute__((ext_vector_type(16))) __bf16 v16bf;
typedef __attribute__((ext_vector_type(8)))  __bf16 v8bf;
typedef __attribute__((ext_vector_type(8)))  float  v8f;
typedef __attribute__((ext_vector_type(4)))  int    v4i;

#define M_DIM 8192
#define N_DIM 11008
#define K_DIM 4096

#define BM 128
#define BN 128
#define BK 32
#define ASTRIDE 40    // LDS row stride for A (ushorts): 80B, 16B-aligned
#define BSTRIDE 136   // LDS row stride for B (ushorts): 272B, 16B-aligned

#if __has_builtin(__builtin_amdgcn_global_load_async_to_lds_b128)
#define HAVE_ASYNC_LDS 1
typedef __attribute__((address_space(1))) v4i GV4;   // global int4
typedef __attribute__((address_space(3))) v4i LV4;   // LDS int4
#endif

struct __align__(16) U8 { unsigned short v[8]; };

__device__ __forceinline__ unsigned short f32_to_bf16_rne(float f) {
  unsigned int u = __float_as_uint(f);
  u += 0x7FFFu + ((u >> 16) & 1u);   // round-to-nearest-even
  return (unsigned short)(u >> 16);
}

__device__ __forceinline__ float wsim_of(float wv, unsigned char mv, float bscale) {
  // mask ? w : sign(w)*scale   (sign(0) == 0, OUTLIER_SCALE == 1)
  return mv ? wv : (wv > 0.f ? bscale : (wv < 0.f ? -bscale : 0.f));
}

// ---------------------------------------------------------------------------
// Phase-1a: x fp32 [M][K] -> bf16 [M][K]
// ---------------------------------------------------------------------------
__global__ __launch_bounds__(256)
void convert_x_bf16(const float* __restrict__ x, unsigned short* __restrict__ xb) {
  const size_t i = ((size_t)blockIdx.x * 256 + threadIdx.x) * 8;
  const float4 a = *(const float4*)(x + i);
  const float4 b = *(const float4*)(x + i + 4);
  U8 h;
  h.v[0] = f32_to_bf16_rne(a.x); h.v[1] = f32_to_bf16_rne(a.y);
  h.v[2] = f32_to_bf16_rne(a.z); h.v[3] = f32_to_bf16_rne(a.w);
  h.v[4] = f32_to_bf16_rne(b.x); h.v[5] = f32_to_bf16_rne(b.y);
  h.v[6] = f32_to_bf16_rne(b.z); h.v[7] = f32_to_bf16_rne(b.w);
  *(U8*)(xb + i) = h;
}

// ---------------------------------------------------------------------------
// Phase-1b: w_sim = (mask ? w : sign(w)*scale) -> bf16, TRANSPOSED to [K][N].
// 64x64 tile per block through LDS: coalesced reads along K, writes along N.
// ---------------------------------------------------------------------------
__global__ __launch_bounds__(256)
void convert_w_bf16T(const float* __restrict__ w,
                     const unsigned char* __restrict__ mask,
                     const float* __restrict__ bscale_p,
                     unsigned short* __restrict__ wbT) {
  __shared__ __align__(16) unsigned short T[64 * 72];   // [k][n], pad 8
  const float bscale = bscale_p[0];
  const int tid   = threadIdx.x;
  const int nBase = blockIdx.x * 64;                    // 172 blocks
  const int kBase = blockIdx.y * 64;                    // 64 blocks

  // load: 4 threads per n-row; each handles 4 float4 along k
  const int n_l = tid >> 2;                             // 0..63
  const int kq  = tid & 3;
#pragma unroll
  for (int p = 0; p < 4; ++p) {
    const int k_l = (kq + p * 4) * 4;                   // 0..60 step 16/4
    const size_t g = (size_t)(nBase + n_l) * K_DIM + kBase + k_l;
    const float4 wv = *(const float4*)(w + g);
    const uchar4 mv = *(const uchar4*)(mask + g);
    T[(k_l + 0) * 72 + n_l] = f32_to_bf16_rne(wsim_of(wv.x, mv.x, bscale));
    T[(k_l + 1) * 72 + n_l] = f32_to_bf16_rne(wsim_of(wv.y, mv.y, bscale));
    T[(k_l + 2) * 72 + n_l] = f32_to_bf16_rne(wsim_of(wv.z, mv.z, bscale));
    T[(k_l + 3) * 72 + n_l] = f32_to_bf16_rne(wsim_of(wv.w, mv.w, bscale));
  }
  __syncthreads();

  // store: 4 threads per k-row; each writes 16 contiguous n (2x16B)
  const int k_l = tid >> 2;                             // 0..63
  const int nq  = (tid & 3) * 16;
  const size_t o = (size_t)(kBase + k_l) * N_DIM + nBase + nq;
  *(U8*)(wbT + o)     = *(const U8*)&T[k_l * 72 + nq];
  *(U8*)(wbT + o + 8) = *(const U8*)&T[k_l * 72 + nq + 8];
}

// ---------------------------------------------------------------------------
// Phase-2: bf16 GEMM. A = xb [M][K], B = wbT [K][N]. 128x128 block tile,
// 8 wave32s in 2x4 grid, each wave 64x32 (4x2 WMMA frags), BK=32,
// double-buffered LDS fed by async global->LDS copies.
// ---------------------------------------------------------------------------
__global__ __launch_bounds__(256, 2)
void gemm_bf16_wmma(const unsigned short* __restrict__ xb,
                    const unsigned short* __restrict__ wbT,
                    const float* __restrict__ bias,
                    float* __restrict__ out) {
  __shared__ __align__(16) unsigned short As[2][BM * ASTRIDE];
  __shared__ __align__(16) unsigned short Bs[2][BK * BSTRIDE];

  const int tid   = threadIdx.x;
  const int lane  = tid & 31;
  const int wave  = tid >> 5;
  const int waveM = wave >> 2;               // 0..1
  const int waveN = wave & 3;                // 0..3
  const int mBlock = blockIdx.y * BM;
  const int nBlock = blockIdx.x * BN;

  // staging map: A: thread -> row m=tid>>1, 32B half (tid&1). B: row k=tid>>3,
  // 32B of n (tid&7). Both give fully coalesced 16B-chunk global reads.
  const int aRow = tid >> 1, aOff = (tid & 1) * 16;     // ushort units
  const int bRow = tid >> 3, bOff = (tid & 7) * 16;
  const unsigned short* gA = xb  + (size_t)(mBlock + aRow) * K_DIM + aOff;
  const unsigned short* gB = wbT + (size_t)bRow * N_DIM + nBlock + bOff;
  const int ldsA = aRow * ASTRIDE + aOff;
  const int ldsB = bRow * BSTRIDE + bOff;

  v8f acc[4][2];
#pragma unroll
  for (int i = 0; i < 4; ++i)
#pragma unroll
    for (int j = 0; j < 2; ++j)
#pragma unroll
      for (int r = 0; r < 8; ++r) acc[i][j][r] = 0.0f;

  const int khalf = lane >> 4;
  const int arow  = lane & 15;

  auto stage = [&](int buf, int kk) {
    const unsigned short* ga = gA + kk;                 // advance along K
    const unsigned short* gb = gB + (size_t)kk * N_DIM; // advance K rows
    unsigned short* la = &As[buf][ldsA];
    unsigned short* lb = &Bs[buf][ldsB];
#ifdef HAVE_ASYNC_LDS
    __builtin_amdgcn_global_load_async_to_lds_b128((GV4*)(ga),     (LV4*)(la),     0, 0);
    __builtin_amdgcn_global_load_async_to_lds_b128((GV4*)(ga + 8), (LV4*)(la + 8), 0, 0);
    __builtin_amdgcn_global_load_async_to_lds_b128((GV4*)(gb),     (LV4*)(lb),     0, 0);
    __builtin_amdgcn_global_load_async_to_lds_b128((GV4*)(gb + 8), (LV4*)(lb + 8), 0, 0);
#else
    ((U8*)la)[0] = ((const U8*)ga)[0];
    ((U8*)la)[1] = ((const U8*)ga)[1];
    ((U8*)lb)[0] = ((const U8*)gb)[0];
    ((U8*)lb)[1] = ((const U8*)gb)[1];
#endif
  };

  const int nIter = K_DIM / BK;                         // 128
  stage(0, 0);
  for (int it = 0; it < nIter; ++it) {
    const int buf = it & 1;
    if (it + 1 < nIter) {
      stage(buf ^ 1, (it + 1) * BK);
#ifdef HAVE_ASYNC_LDS
#if __has_builtin(__builtin_amdgcn_s_wait_asynccnt)
      __builtin_amdgcn_s_wait_asynccnt(4);              // stage(it) landed
#else
      asm volatile("s_wait_asynccnt 0x4" ::: "memory");
#endif
#endif
    } else {
#ifdef HAVE_ASYNC_LDS
#if __has_builtin(__builtin_amdgcn_s_wait_asynccnt)
      __builtin_amdgcn_s_wait_asynccnt(0);
#else
      asm volatile("s_wait_asynccnt 0x0" ::: "memory");
#endif
#endif
    }
    __syncthreads();

    // ---- WMMA: 4 A-frags x 2 B-frags per wave ----
    v16bf a[4], b[2];
#pragma unroll
    for (int i = 0; i < 4; ++i) {
      // A 16x32 bf16: lane(0-15)=M; VGPR0-3 K[kh*8..+7], VGPR4-7 K[16+kh*8..+7]
      const unsigned short* ap =
          &As[buf][(waveM * 64 + i * 16 + arow) * ASTRIDE + khalf * 8];
      union { v16bf v; v8bf h[2]; } u;
      u.h[0] = *(const v8bf*)(ap);
      u.h[1] = *(const v8bf*)(ap + 16);
      a[i] = u.v;
    }
#pragma unroll
    for (int j = 0; j < 2; ++j) {
      // B 32x16 bf16: lane(0-15)=N; elems 0-7 K[kh*8..+7], 8-15 K[16+kh*8..+7]
      const unsigned short* bp = &Bs[buf][(khalf * 8) * BSTRIDE + waveN * 32 + j * 16 + arow];
      union { v16bf v; unsigned short s[16]; } u;
#pragma unroll
      for (int s = 0; s < 8; ++s) {
        u.s[s]     = bp[s * BSTRIDE];
        u.s[s + 8] = bp[(16 + s) * BSTRIDE];
      }
      b[j] = u.v;
    }
#pragma unroll
    for (int i = 0; i < 4; ++i)
#pragma unroll
      for (int j = 0; j < 2; ++j)
        acc[i][j] = __builtin_amdgcn_wmma_f32_16x16x32_bf16(
            false, a[i], false, b[j], (short)0, acc[i][j], false, false);
    __syncthreads();
  }

  // ---- epilogue: D VGPR r -> M = r + 8*(lane>=16), N = lane&15 ----
  const int lhalf = lane >> 4;
  const int lcol  = lane & 15;
#pragma unroll
  for (int j = 0; j < 2; ++j) {
    const int col = nBlock + waveN * 32 + j * 16 + lcol;
    const float bz = bias[col];
#pragma unroll
    for (int i = 0; i < 4; ++i) {
      const int rowBase = mBlock + waveM * 64 + i * 16 + lhalf * 8;
#pragma unroll
      for (int r = 0; r < 8; ++r)
        out[(size_t)(rowBase + r) * N_DIM + col] = acc[i][j][r] + bz;
    }
  }
}

// ---------------------------------------------------------------------------
// Fallback: fused single kernel (no workspace needed)
// ---------------------------------------------------------------------------
__global__ __launch_bounds__(256, 2)
void binxnor_linear_fused(const float* __restrict__ x,
                          const float* __restrict__ w,
                          const float* __restrict__ bias,
                          const unsigned char* __restrict__ mask,
                          const float* __restrict__ bscale_p,
                          float* __restrict__ out) {
  __shared__ __align__(16) unsigned short As[BM * ASTRIDE];
  __shared__ __align__(16) unsigned short Bs[BK * BSTRIDE];

  const float bscale = bscale_p[0];
  const int tid   = threadIdx.x;
  const int lane  = tid & 31;
  const int wave  = tid >> 5;
  const int waveM = wave >> 2;
  const int waveN = wave & 3;
  const int mBlock = blockIdx.y * BM;
  const int nBlock = blockIdx.x * BN;
  const int rowsel = tid >> 3;
  const int k4     = (tid & 7) * 4;

  v8f acc[4][2];
#pragma unroll
  for (int i = 0; i < 4; ++i)
#pragma unroll
    for (int j = 0; j < 2; ++j)
#pragma unroll
      for (int r = 0; r < 8; ++r) acc[i][j][r] = 0.0f;

  const int khalf = lane >> 4;
  const int arow  = lane & 15;

  for (int kk = 0; kk < K_DIM; kk += BK) {
#pragma unroll
    for (int p = 0; p < 4; ++p) {
      const int m = rowsel + p * 32;
      const float4 xv = *(const float4*)(x + (size_t)(mBlock + m) * K_DIM + kk + k4);
      ushort4 h;
      h.x = f32_to_bf16_rne(xv.x); h.y = f32_to_bf16_rne(xv.y);
      h.z = f32_to_bf16_rne(xv.z); h.w = f32_to_bf16_rne(xv.w);
      *(ushort4*)&As[m * ASTRIDE + k4] = h;
    }
#pragma unroll
    for (int p = 0; p < 4; ++p) {
      const int n = rowsel + p * 32;
      const size_t g = (size_t)(nBlock + n) * K_DIM + kk + k4;
      const float4 wv = *(const float4*)(w + g);
      const uchar4 mv = *(const uchar4*)(mask + g);
      Bs[(k4 + 0) * BSTRIDE + n] = f32_to_bf16_rne(wsim_of(wv.x, mv.x, bscale));
      Bs[(k4 + 1) * BSTRIDE + n] = f32_to_bf16_rne(wsim_of(wv.y, mv.y, bscale));
      Bs[(k4 + 2) * BSTRIDE + n] = f32_to_bf16_rne(wsim_of(wv.z, mv.z, bscale));
      Bs[(k4 + 3) * BSTRIDE + n] = f32_to_bf16_rne(wsim_of(wv.w, mv.w, bscale));
    }
    __syncthreads();

    v16bf a[4], b[2];
#pragma unroll
    for (int i = 0; i < 4; ++i) {
      const unsigned short* ap =
          &As[(waveM * 64 + i * 16 + arow) * ASTRIDE + khalf * 8];
      union { v16bf v; v8bf h[2]; } u;
      u.h[0] = *(const v8bf*)(ap);
      u.h[1] = *(const v8bf*)(ap + 16);
      a[i] = u.v;
    }
#pragma unroll
    for (int j = 0; j < 2; ++j) {
      const unsigned short* bp = &Bs[(khalf * 8) * BSTRIDE + waveN * 32 + j * 16 + arow];
      union { v16bf v; unsigned short s[16]; } u;
#pragma unroll
      for (int s = 0; s < 8; ++s) {
        u.s[s]     = bp[s * BSTRIDE];
        u.s[s + 8] = bp[(16 + s) * BSTRIDE];
      }
      b[j] = u.v;
    }
#pragma unroll
    for (int i = 0; i < 4; ++i)
#pragma unroll
      for (int j = 0; j < 2; ++j)
        acc[i][j] = __builtin_amdgcn_wmma_f32_16x16x32_bf16(
            false, a[i], false, b[j], (short)0, acc[i][j], false, false);
    __syncthreads();
  }

  const int lhalf = lane >> 4;
  const int lcol  = lane & 15;
#pragma unroll
  for (int j = 0; j < 2; ++j) {
    const int col = nBlock + waveN * 32 + j * 16 + lcol;
    const float bz = bias[col];
#pragma unroll
    for (int i = 0; i < 4; ++i) {
      const int rowBase = mBlock + waveM * 64 + i * 16 + lhalf * 8;
#pragma unroll
      for (int r = 0; r < 8; ++r)
        out[(size_t)(rowBase + r) * N_DIM + col] = acc[i][j][r] + bz;
    }
  }
}

extern "C" void kernel_launch(void* const* d_in, const int* in_sizes, int n_in,
                              void* d_out, int out_size, void* d_ws, size_t ws_size,
                              hipStream_t stream) {
  (void)in_sizes; (void)n_in; (void)out_size;
  const float*         x      = (const float*)d_in[0];         // [8192, 4096]
  const float*         w      = (const float*)d_in[1];         // [11008, 4096]
  const float*         bias   = (const float*)d_in[2];         // [11008]
  const unsigned char* mask   = (const unsigned char*)d_in[3]; // [11008, 4096] bool
  const float*         bscale = (const float*)d_in[4];         // [1]
  float*               out    = (float*)d_out;                 // [8192, 11008]

  const size_t needA = (size_t)M_DIM * K_DIM * sizeof(unsigned short); // 67 MB
  const size_t needB = (size_t)K_DIM * N_DIM * sizeof(unsigned short); // 90 MB
  dim3 gGemm(N_DIM / BN, M_DIM / BM);   // (86, 64)

  if (d_ws && ws_size >= needA + needB) {
    unsigned short* xb  = (unsigned short*)d_ws;
    unsigned short* wbT = (unsigned short*)((char*)d_ws + needA);
    convert_x_bf16<<<(M_DIM * (size_t)K_DIM / 8) / 256, 256, 0, stream>>>(x, xb);
    convert_w_bf16T<<<dim3(N_DIM / 64, K_DIM / 64), 256, 0, stream>>>(w, mask, bscale, wbT);
    gemm_bf16_wmma<<<gGemm, 256, 0, stream>>>(xb, wbT, bias, out);
  } else {
    binxnor_linear_fused<<<gGemm, 256, 0, stream>>>(x, w, bias, mask, bscale, out);
  }
}
